// T5LayerSelfAttention_11708080849456
// MI455X (gfx1250) — compile-verified, matrix operand order checked by reading, not run on hardware
//
#include <hip/hip_runtime.h>

// ---------------------------------------------------------------------------
// T5 self-attention layer for gfx1250 (MI455X).
// f16 WMMA w/ f32 accumulate. Tile staging:
//   - GEMMs: Tensor Data Mover (tensor_load_to_lds, TENSORcnt) w/ async fallback
//   - attention: per-lane global_load_async_to_lds_b128 (ASYNCcnt)
// Both double-buffered.
// ---------------------------------------------------------------------------

typedef __attribute__((ext_vector_type(16))) _Float16 v16h;
typedef __attribute__((ext_vector_type(8)))  _Float16 v8h;
typedef __attribute__((ext_vector_type(8)))  float    v8f;
typedef __attribute__((ext_vector_type(4)))  unsigned u32x4;
typedef __attribute__((ext_vector_type(8)))  int      i32x8;
typedef __attribute__((ext_vector_type(4)))  int      i32x4;

#define HH   16          // heads
#define DKV  64          // head dim
#define SEQ  2048
#define DIM  1024        // model dim == HH*DKV
#define BB   2
#define MTOT (BB * SEQ)  // 4096 rows

#if defined(__AMDGCN__) && __has_builtin(__builtin_amdgcn_tensor_load_to_lds)
#define HAVE_TDM 1
#else
#define HAVE_TDM 0
#endif

static __device__ __forceinline__ v16h make_v16(v8h lo, v8h hi) {
  union { v16h v; v8h h[2]; } u;
  u.h[0] = lo; u.h[1] = hi;
  return u.v;
}

static __device__ __forceinline__ v8f wmma16(v16h a, v16h b, v8f c) {
  // D = A(16x32 f16) x B(32x16 f16) + C(16x16 f32)
  return __builtin_amdgcn_wmma_f32_16x16x32_f16(
      /*neg_a=*/false, a, /*neg_b=*/false, b,
      /*c_mod=*/(short)0, c, /*reuse_a=*/false, /*reuse_b=*/false);
}

// 16-byte async copy global -> LDS (per lane). Tracked by ASYNCcnt, no VGPR
// data movement. Generic shared-pointer low 32 bits == wave LDS byte offset.
static __device__ __forceinline__ void async_b128(void* lds, const void* g) {
  asm volatile("global_load_async_to_lds_b128 %0, %1, off"
               :: "v"((unsigned)(unsigned long long)lds),
                  "v"((unsigned long long)g)
               : "memory");
}
static __device__ __forceinline__ void wait_async_le4() {
  asm volatile("s_wait_asynccnt 0x4" ::: "memory");
}
static __device__ __forceinline__ void wait_async_0() {
  asm volatile("s_wait_asynccnt 0x0" ::: "memory");
}

#if HAVE_TDM
// One TDM DMA of a 2-D tile (tile_x elems of 2 bytes per row, tile_y rows,
// row pitch stride_elems) from a row-major f16 tensor into LDS (row-major,
// contiguous). D# per CDNA5 ISA 8.3/8.4; groups 2/3 unused for 2-D tensors.
// This toolchain exposes the 6-arg builtin:
//   (u32x4 g0, i32x8 g1, i32x4 g2, i32x4 g3, i32x8 extra, i32 cpol)
static __device__ __forceinline__ void tdm_load_2d(void* lds, const void* gsrc,
                                                   unsigned tile_x, unsigned tile_y,
                                                   unsigned stride_elems) {
  const unsigned long long ga = (unsigned long long)gsrc;
  u32x4 g0;
  g0.x = 1u;                                              // count=1, user D#
  g0.y = (unsigned)(unsigned long long)lds;               // lds_addr (bytes)
  g0.z = (unsigned)(ga & 0xffffffffu);                    // global_addr[31:0]
  g0.w = (unsigned)((ga >> 32) & 0x01ffffffu) | (2u << 30); // addr[56:32]|type=2
  const unsigned dim0 = stride_elems;                     // in-bounds by constr.
  const unsigned dim1 = 1u << 20;
  i32x8 g1;
  g1[0] = 0x10000;                       // workgroup_mask=0, data_size=1 (2 B)
  g1[1] = (int)((dim0 & 0xffffu) << 16);                  // tensor_dim0 lo16
  g1[2] = (int)((dim0 >> 16) | ((dim1 & 0xffffu) << 16)); // dim0 hi16 | dim1 lo16
  g1[3] = (int)((dim1 >> 16) | (tile_x << 16));           // dim1 hi16 | tile_dim0
  g1[4] = (int)(tile_y & 0xffffu);                        // tile_dim1 (tile_dim2=0)
  g1[5] = (int)stride_elems;                              // tensor_dim0_stride lo32
  g1[6] = 0;                                              // stride hi16 | dim1_stride lo
  g1[7] = 0;
  const i32x4 gz4 = {0, 0, 0, 0};
  const i32x8 gz8 = {0, 0, 0, 0, 0, 0, 0, 0};
  __builtin_amdgcn_tensor_load_to_lds(g0, g1, gz4, gz4, gz8, 0);
}
#endif

// ---------------------------------------------------------------------------
// Kernel 1: RMS norm (T5 LayerNorm: no mean subtraction) -> f16 activations
// ---------------------------------------------------------------------------
__global__ void t5_rmsnorm_kernel(const float* __restrict__ hs,
                                  const float* __restrict__ w,
                                  _Float16* __restrict__ out) {
  __shared__ float red[256];
  const int row = blockIdx.x;
  const int t = threadIdx.x;
  const float* r = hs + (size_t)row * DIM;

  float vals[4];
  float s = 0.f;
#pragma unroll
  for (int i = 0; i < 4; ++i) {
    vals[i] = r[t + i * 256];
    s += vals[i] * vals[i];
  }
  red[t] = s;
  __syncthreads();
  for (int off = 128; off > 0; off >>= 1) {
    if (t < off) red[t] += red[t + off];
    __syncthreads();
  }
  const float rs = rsqrtf(red[0] * (1.0f / DIM) + 1e-6f);
#pragma unroll
  for (int i = 0; i < 4; ++i) {
    const int c = t + i * 256;
    out[(size_t)row * DIM + c] = (_Float16)(vals[i] * w[c] * rs);
  }
}

// ---------------------------------------------------------------------------
// Kernel 2: convert weights f32 -> f16 AND transpose to [N][K]
// out layout: wqT | wkT | wvT | woT, each [1024][1024] f16
// ---------------------------------------------------------------------------
__global__ void t5_convw_kernel(const float* __restrict__ wq,
                                const float* __restrict__ wk,
                                const float* __restrict__ wv,
                                const float* __restrict__ wo,
                                _Float16* __restrict__ out) {
  const int id = blockIdx.x * 256 + threadIdx.x;     // 0 .. 4M-1
  const int w = id >> 20;
  const int rem = id & ((1 << 20) - 1);
  const int e = rem >> 10;       // output feature (row of transposed)
  const int d = rem & 1023;      // input feature  (col of transposed)
  const float* src = (w == 0) ? wq : (w == 1) ? wk : (w == 2) ? wv : wo;
  out[((size_t)w << 20) + (size_t)e * 1024 + d] = (_Float16)src[(size_t)d * 1024 + e];
}

// ---------------------------------------------------------------------------
// Kernel 3/5: tiled GEMM, C[M,N] = A[M,K] * Bt[N,K]^T  (M=4096, N=K=1024)
// Block tile 128x128, K-step 32. Tiles staged by the Tensor Data Mover
// (one 128x32 f16 descriptor each for A and B, issued by wave 0, TENSORcnt
// ping-pong) or per-lane async copies when TDM builtin is unavailable.
//   mode 0: Q  -> [b,h,s,64] f16       mode 1: K -> [b,h,s,64] f16
//   mode 2: V  -> [b,h,64,s] f16       mode 3: f32 out = hidden + acc
// ---------------------------------------------------------------------------
__global__ void t5_gemm_kernel(const _Float16* __restrict__ A,
                               const _Float16* __restrict__ Bt,
                               int mode_base,
                               _Float16* __restrict__ q16,
                               _Float16* __restrict__ k16,
                               _Float16* __restrict__ vt16,
                               const float* __restrict__ hidden,
                               float* __restrict__ outf) {
  __shared__ _Float16 sA[2][128 * 32];   // 2 x 8 KB
  __shared__ _Float16 sB[2][128 * 32];   // 2 x 8 KB

  const int mode = mode_base + blockIdx.z;
  const _Float16* B = Bt + ((size_t)blockIdx.z << 20);
  const int m0 = blockIdx.y * 128;
  const int n0 = blockIdx.x * 128;
  const int t = threadIdx.x;
  const int lane = t & 31, wave = t >> 5;
  const int wm = wave >> 2, wn = wave & 3;
  const int ln = lane & 15, hi = lane >> 4;

  const v8f zero = {0.f, 0.f, 0.f, 0.f, 0.f, 0.f, 0.f, 0.f};
  v8f acc[4][2];
#pragma unroll
  for (int i = 0; i < 4; ++i)
#pragma unroll
    for (int j = 0; j < 2; ++j) acc[i][j] = zero;

#if HAVE_TDM
  auto stage = [&](int k0, int buf) {
    if (wave == 0) {
      tdm_load_2d(&sA[buf][0], &A[(size_t)m0 * 1024 + k0], 32, 128, 1024);
      tdm_load_2d(&sB[buf][0], &B[(size_t)n0 * 1024 + k0], 32, 128, 1024);
    }
  };
  auto wait_stage = [&](bool pending) {
    if (wave == 0) {
      if (pending) __builtin_amdgcn_s_wait_tensorcnt(2);
      else         __builtin_amdgcn_s_wait_tensorcnt(0);
    }
  };
#else
  auto stage = [&](int k0, int buf) {
#pragma unroll
    for (int i = 0; i < 2; ++i) {
      const int c = t + i * 256;               // 0..511 chunk id
      const int row = c >> 2, col8 = (c & 3) * 8;
      async_b128(&sA[buf][row * 32 + col8],
                 &A[(size_t)(m0 + row) * 1024 + k0 + col8]);
      async_b128(&sB[buf][row * 32 + col8],
                 &B[(size_t)(n0 + row) * 1024 + k0 + col8]);
    }
  };
  auto wait_stage = [&](bool pending) {
    if (pending) wait_async_le4();
    else         wait_async_0();
  };
#endif

  const int nk = 1024 / 32;
  stage(0, 0);
  for (int ik = 0; ik < nk; ++ik) {
    const int cur = ik & 1;
    const bool pre = (ik + 1 < nk);
    if (pre) stage((ik + 1) * 32, cur ^ 1);  // prefetch next tile
    wait_stage(pre);                         // older DMAs (current buf) landed
    __syncthreads();                         // current buffer visible block-wide

    v16h af[4], bf[2];
#pragma unroll
    for (int i = 0; i < 4; ++i) {
      const int r = wm * 64 + i * 16 + ln;
      af[i] = make_v16(*(const v8h*)&sA[cur][r * 32 + hi * 8],
                       *(const v8h*)&sA[cur][r * 32 + hi * 8 + 16]);
    }
#pragma unroll
    for (int j = 0; j < 2; ++j) {
      const int r = wn * 32 + j * 16 + ln;
      bf[j] = make_v16(*(const v8h*)&sB[cur][r * 32 + hi * 8],
                       *(const v8h*)&sB[cur][r * 32 + hi * 8 + 16]);
    }
#pragma unroll
    for (int i = 0; i < 4; ++i)
#pragma unroll
      for (int j = 0; j < 2; ++j) acc[i][j] = wmma16(af[i], bf[j], acc[i][j]);

    __syncthreads();          // all waves done reading -> buffer reusable
  }

  // epilogue: C layout is lane(0..15)=N, vgpr r = M (lanes>=16 -> M=r+8)
#pragma unroll
  for (int i = 0; i < 4; ++i)
#pragma unroll
    for (int j = 0; j < 2; ++j) {
      const int mbase = m0 + wm * 64 + i * 16 + hi * 8;
      const int nc = n0 + wn * 32 + j * 16 + ln;
#pragma unroll
      for (int r = 0; r < 8; ++r) {
        const int m = mbase + r;
        const float v = acc[i][j][r];
        if (mode == 3) {
          outf[(size_t)m * 1024 + nc] = hidden[(size_t)m * 1024 + nc] + v;
        } else {
          const int b = m >> 11, s = m & 2047;
          const int h = nc >> 6, d = nc & 63;
          if (mode == 0)
            q16[((size_t)(b * HH + h) * SEQ + s) * DKV + d] = (_Float16)v;
          else if (mode == 1)
            k16[((size_t)(b * HH + h) * SEQ + s) * DKV + d] = (_Float16)v;
          else
            vt16[((size_t)(b * HH + h) * DKV + d) * SEQ + s] = (_Float16)v;
        }
      }
    }
}

// ---------------------------------------------------------------------------
// Kernel 4: flash attention per (b,h). Block = 8 waves, each wave owns 16
// q-rows (block covers 128). K/V streamed in 64-wide chunks through
// double-buffered LDS tiles filled with per-lane async copies (ASYNCcnt).
// ---------------------------------------------------------------------------
__global__ void t5_attn_kernel(const _Float16* __restrict__ q16,
                               const _Float16* __restrict__ k16,
                               const _Float16* __restrict__ vt16,
                               const float* __restrict__ rel_bias,
                               _Float16* __restrict__ ctx16) {
  __shared__ _Float16 biasH[4096];        // bias[k - q + 2047]      8 KB
  __shared__ _Float16 kt[2][64 * 64];     // K chunk [k_row][d]   2x 8 KB
  __shared__ _Float16 vt[2][64 * 64];     // V chunk [d][k_col]   2x 8 KB
  __shared__ _Float16 pbuf[8 * 16 * 64];  // per-wave P tiles       16 KB

  const int bh = blockIdx.y;             // b*16 + h
  const int h = bh & 15;
  const int b = bh >> 4;
  const int t = threadIdx.x, lane = t & 31, wave = t >> 5;
  const int ln = lane & 15, hi = lane >> 4;
  const int q0 = blockIdx.x * 128 + wave * 16;

  const _Float16* qbase = q16 + (size_t)bh * SEQ * DKV;
  const _Float16* kbase = k16 + (size_t)bh * SEQ * DKV;
  const _Float16* vbase = vt16 + (size_t)bh * DKV * SEQ;

  // stage one 64-wide K/V chunk (4 async b128 per thread)
  auto stage = [&](int kc, int buf) {
#pragma unroll
    for (int i = 0; i < 2; ++i) {
      const int c = t + i * 256;                    // 0..511 chunks of 8 halfs
      const int row = c >> 3, col8 = (c & 7) * 8;
      async_b128(&kt[buf][row * 64 + col8],
                 &kbase[(size_t)(kc + row) * DKV + col8]);
      async_b128(&vt[buf][row * 64 + col8],
                 &vbase[(size_t)row * SEQ + kc + col8]);
    }
  };
  stage(0, 0);

  // T5 bidirectional bucket: 16 sign buckets x {exact<8 | log-spaced to 128}
  for (int i = t; i < 4095; i += 256) {
    const int delta = i - 2047;          // k - q
    const int rb = (delta > 0) ? 16 : 0;
    const int rp = (delta < 0) ? -delta : delta;
    int bu;
    if (rp < 8) {
      bu = rp;
    } else {
      bu = 8 + (int)(__logf((float)rp * 0.125f) * (8.0f / __logf(16.0f)));
      bu = bu > 15 ? 15 : bu;
    }
    biasH[i] = (_Float16)rel_bias[(rb + bu) * HH + h];
  }

  // Q fragments for this wave's 16 rows (kept in VGPRs for the whole kernel)
  v16h qf[2];
  {
    const _Float16* p = qbase + (size_t)(q0 + ln) * DKV;
    qf[0] = make_v16(*(const v8h*)&p[hi * 8], *(const v8h*)&p[hi * 8 + 16]);
    qf[1] = make_v16(*(const v8h*)&p[32 + hi * 8], *(const v8h*)&p[32 + hi * 8 + 16]);
  }

  const v8f zero = {0.f, 0.f, 0.f, 0.f, 0.f, 0.f, 0.f, 0.f};
  float m_i[8], l_i[8];
  v8f o[4];
#pragma unroll
  for (int r = 0; r < 8; ++r) { m_i[r] = -1e30f; l_i[r] = 0.f; }
#pragma unroll
  for (int j = 0; j < 4; ++j) o[j] = zero;

  _Float16* myp = &pbuf[wave * 16 * 64];

  const int nchunk = SEQ / 64;
  for (int ic = 0; ic < nchunk; ++ic) {
    const int kc = ic * 64;
    const int cur = ic & 1;
    if (ic + 1 < nchunk) {      // prefetch next K/V chunk
      stage(kc + 64, cur ^ 1);
      wait_async_le4();
    } else {
      wait_async_0();
    }
    __syncthreads();            // current K/V tiles (+ bias table) visible

    // scores S = Q(16x64) * K_chunk^T(64x64)
    v8f s[4];
#pragma unroll
    for (int ns = 0; ns < 4; ++ns) {
      const int r = ns * 16 + ln;
      v16h b0 = make_v16(*(const v8h*)&kt[cur][r * 64 + hi * 8],
                         *(const v8h*)&kt[cur][r * 64 + hi * 8 + 16]);
      v16h b1 = make_v16(*(const v8h*)&kt[cur][r * 64 + 32 + hi * 8],
                         *(const v8h*)&kt[cur][r * 64 + 32 + hi * 8 + 16]);
      v8f a = zero;
      a = wmma16(qf[0], b0, a);
      a = wmma16(qf[1], b1, a);
      s[ns] = a;
    }

    // bias add + per-row max (rows live across 16-lane halves in wave32)
    float cm[8];
#pragma unroll
    for (int r = 0; r < 8; ++r) {
      const int qrow = q0 + r + hi * 8;
      float mx = -1e30f;
#pragma unroll
      for (int ns = 0; ns < 4; ++ns) {
        const int kcol = kc + ns * 16 + ln;
        const float v = s[ns][r] + (float)biasH[kcol - qrow + 2047];
        s[ns][r] = v;
        mx = fmaxf(mx, v);
      }
      cm[r] = mx;
    }
#pragma unroll
    for (int msk = 1; msk < 16; msk <<= 1)
#pragma unroll
      for (int r = 0; r < 8; ++r) cm[r] = fmaxf(cm[r], __shfl_xor(cm[r], msk, 32));

    // online softmax update
    float rs[8];
#pragma unroll
    for (int r = 0; r < 8; ++r) {
      const float mnew = fmaxf(m_i[r], cm[r]);
      const float scale = __expf(m_i[r] - mnew);
      float sum = 0.f;
#pragma unroll
      for (int ns = 0; ns < 4; ++ns) {
        const float p = __expf(s[ns][r] - mnew);
        s[ns][r] = p;
        sum += p;
      }
      rs[r] = sum;
      m_i[r] = mnew;
      l_i[r] *= scale;
#pragma unroll
      for (int j = 0; j < 4; ++j) o[j][r] *= scale;
    }
#pragma unroll
    for (int msk = 1; msk < 16; msk <<= 1)
#pragma unroll
      for (int r = 0; r < 8; ++r) rs[r] += __shfl_xor(rs[r], msk, 32);
#pragma unroll
    for (int r = 0; r < 8; ++r) l_i[r] += rs[r];

    // transpose P from C-layout to A-layout via per-wave LDS tile
#pragma unroll
    for (int ns = 0; ns < 4; ++ns)
#pragma unroll
      for (int r = 0; r < 8; ++r)
        myp[(r + hi * 8) * 64 + ns * 16 + ln] = (_Float16)s[ns][r];
    __syncthreads();            // P visible (and ds writes retired)

    // ctx += P(16x64) * V_chunk(64x64);  Vt LDS tile is already [N=d][K=s]
    v16h pf0 = make_v16(*(const v8h*)&myp[ln * 64 + hi * 8],
                        *(const v8h*)&myp[ln * 64 + hi * 8 + 16]);
    v16h pf1 = make_v16(*(const v8h*)&myp[ln * 64 + 32 + hi * 8],
                        *(const v8h*)&myp[ln * 64 + 32 + hi * 8 + 16]);
#pragma unroll
    for (int ns = 0; ns < 4; ++ns) {
      const int r = ns * 16 + ln;
      v16h b0 = make_v16(*(const v8h*)&vt[cur][r * 64 + hi * 8],
                         *(const v8h*)&vt[cur][r * 64 + hi * 8 + 16]);
      v16h b1 = make_v16(*(const v8h*)&vt[cur][r * 64 + 32 + hi * 8],
                         *(const v8h*)&vt[cur][r * 64 + 32 + hi * 8 + 16]);
      o[ns] = wmma16(pf0, b0, o[ns]);
      o[ns] = wmma16(pf1, b1, o[ns]);
    }
    __syncthreads();            // all waves done reading current buffers
  }

  // normalize and store ctx as [b, s, h*64+d] f16 (rows for out-projection)
#pragma unroll
  for (int r = 0; r < 8; ++r) {
    const float inv = 1.f / l_i[r];
    const int srow = q0 + r + hi * 8;
#pragma unroll
    for (int ns = 0; ns < 4; ++ns) {
      const float v = o[ns][r] * inv;
      ctx16[((size_t)(b * SEQ + srow)) * DIM + h * 64 + ns * 16 + ln] = (_Float16)v;
    }
  }
}

// ---------------------------------------------------------------------------
// host launcher
// ---------------------------------------------------------------------------
extern "C" void kernel_launch(void* const* d_in, const int* in_sizes, int n_in,
                              void* d_out, int out_size, void* d_ws, size_t ws_size,
                              hipStream_t stream) {
  (void)in_sizes; (void)n_in; (void)out_size; (void)ws_size;
  const float* hidden = (const float*)d_in[0];
  const float* lnw    = (const float*)d_in[1];
  const float* wq     = (const float*)d_in[2];
  const float* wk     = (const float*)d_in[3];
  const float* wv     = (const float*)d_in[4];
  const float* wo     = (const float*)d_in[5];
  const float* rb     = (const float*)d_in[6];
  float* out = (float*)d_out;

  char* ws = (char*)d_ws;
  const size_t MB = 1024ull * 1024ull;
  _Float16* normed = (_Float16*)(ws);             //  8 MB  [4096][1024]
  _Float16* w16    = (_Float16*)(ws + 8  * MB);   //  8 MB  wqT|wkT|wvT|woT
  _Float16* q16    = (_Float16*)(ws + 16 * MB);   //  8 MB  [b,h,s,64]
  _Float16* k16    = (_Float16*)(ws + 24 * MB);   //  8 MB  [b,h,s,64]
  _Float16* vt16   = (_Float16*)(ws + 32 * MB);   //  8 MB  [b,h,64,s]
  _Float16* ctx16  = (_Float16*)(ws + 40 * MB);   //  8 MB  [b*s][1024]

  t5_rmsnorm_kernel<<<MTOT, 256, 0, stream>>>(hidden, lnw, normed);
  t5_convw_kernel<<<16384, 256, 0, stream>>>(wq, wk, wv, wo, w16);
  // Q, K, Vt projections: grid.z selects the weight matrix + store layout
  t5_gemm_kernel<<<dim3(8, 32, 3), 256, 0, stream>>>(
      normed, w16, /*mode_base=*/0, q16, k16, vt16, nullptr, nullptr);
  t5_attn_kernel<<<dim3(SEQ / 128, BB * HH), 256, 0, stream>>>(
      q16, k16, vt16, rb, ctx16);
  // output projection + residual (mode 3)
  t5_gemm_kernel<<<dim3(8, 32, 1), 256, 0, stream>>>(
      ctx16, w16 + 3ull * (1u << 20), /*mode_base=*/3, nullptr, nullptr, nullptr,
      hidden, out);
}
